// QComplexTransformerEncoderLayer_78254304133586
// MI455X (gfx1250) — matrix operand order repaired and checked by reading, not compiled, not run on hardware
//
#include <hip/hip_runtime.h>
#include <hip/hip_bf16.h>
#include <math.h>

// ---------------------------------------------------------------------------
// QComplex Transformer Encoder Layer for MI455X (gfx1250, wave32, WMMA)
//   B=4, S=2048, EMBED=1024, FFN=4096, fp32 in/out, bf16 WMMA internally.
// Compute-bound (~1.24 TFLOP vs ~1.5GB traffic): everything heavy runs on
// v_wmma_f32_16x16x32_bf16. 192MB L2 holds all weights -> direct global
// fragment loads with contiguous k per lane half (k = 16*(lane/16)+s) + global_prefetch_b8.
// ---------------------------------------------------------------------------

typedef __attribute__((ext_vector_type(16))) __bf16 v16bf;
typedef __attribute__((ext_vector_type(8)))  __bf16 v8bf;
typedef __attribute__((ext_vector_type(8)))  float  v8f;
typedef __bf16 bf16;

__device__ __forceinline__ unsigned short f32_bf16(float f) {
  unsigned u = __float_as_uint(f);
  u += 0x7FFFu + ((u >> 16) & 1u);          // round-to-nearest-even
  return (unsigned short)(u >> 16);
}

// A-matrix 16x32 bf16 fragment: lanes 0-15 hold K=[0..15],
// lanes 16-31 hold K=[16..31] of row M=lane%16. Two 16B contiguous loads.
__device__ __forceinline__ v16bf load_a_frag(const bf16* p) {
  v8bf lo = *(const v8bf*)(p);
  v8bf hi = *(const v8bf*)(p + 8);
  v16bf r;
#pragma unroll
  for (int i = 0; i < 8; ++i) { r[i] = lo[i]; r[i + 8] = hi[i]; }
  return r;
}

// B-matrix 32x16 fragment from K-major (N x K row-major, i.e. weight (dout,din))
// storage: lane holds column n=lane%16, K=(lane/16)*16 + [0..15]: one 32B load.
// For K-major-in-rows (non-transposed V), use the strided gather below.
__device__ __forceinline__ v16bf load_b_strided(const bf16* p, int ld) {
  v16bf r;
#pragma unroll
  for (int i = 0; i < 16; ++i) r[i] = p[(size_t)i * (size_t)ld];
  return r;
}

__device__ __forceinline__ v8f wmma_bf16(v16bf a, v16bf b, v8f c) {
  return __builtin_amdgcn_wmma_f32_16x16x32_bf16(
      /*neg_a=*/false, a, /*neg_b=*/false, b, /*c_mod=*/(short)0, c,
      /*reuse_a=*/false, /*reuse_b=*/false);
}

// -------- block reductions (256 threads = 8 wave32s) -----------------------
__device__ __forceinline__ float bsum(float v, float* sred) {
#pragma unroll
  for (int o = 16; o > 0; o >>= 1) v += __shfl_xor(v, o, 32);
  if ((threadIdx.x & 31) == 0) sred[threadIdx.x >> 5] = v;
  __syncthreads();
  float r = sred[0];
#pragma unroll
  for (int i = 1; i < 8; ++i) r += sred[i];
  __syncthreads();
  return r;
}
__device__ __forceinline__ float bmax(float v, float* sred) {
#pragma unroll
  for (int o = 16; o > 0; o >>= 1) v = fmaxf(v, __shfl_xor(v, o, 32));
  if ((threadIdx.x & 31) == 0) sred[threadIdx.x >> 5] = v;
  __syncthreads();
  float r = sred[0];
#pragma unroll
  for (int i = 1; i < 8; ++i) r = fmaxf(r, sred[i]);
  __syncthreads();
  return r;
}

// ---------------------------------------------------------------------------
// Fused QComplex GEMM: per 16x16 tile keep 4 f32 accumulators and issue 4
// WMMAs per K-step over 2 shared A-frags and 2 shared B-frags:
//   oa = Aa.Ba^T + coef*Ab.Bb^T + bias_a ; ob = Aa.Bb^T + Ab.Ba^T + bias_b
// mode: 0=f32 out, 1=bf16 out, 2=exact-GELU+bf16 out, 3=magnitude f32 (outA)
// Block = 256 thr (8 waves) covering a 64x32 macro-tile. Dims divide tiles.
// ---------------------------------------------------------------------------
__global__ __launch_bounds__(256) void qc_gemm_bf16(
    const bf16* __restrict__ Aa, const bf16* __restrict__ Ab,
    const bf16* __restrict__ Ba, const bf16* __restrict__ Bb,
    const float* __restrict__ bias_a, const float* __restrict__ bias_b,
    void* __restrict__ outA, void* __restrict__ outB,
    int M, int N, int K,
    long long sA, long long sB, long long sO,
    float coef, float oscale, int mode) {
  const int lane = threadIdx.x & 31;
  const int wave = threadIdx.x >> 5;
  const int m0 = blockIdx.y * 64 + (wave & 3) * 16;
  const int n0 = blockIdx.x * 32 + (wave >> 2) * 16;
  if (m0 >= M || n0 >= N) return;  // wave-uniform; never taken for our dims

  const size_t zA = (size_t)blockIdx.z * (size_t)sA;
  const size_t zB = (size_t)blockIdx.z * (size_t)sB;
  const size_t zO = (size_t)blockIdx.z * (size_t)sO;

  const bf16* rAa = Aa + zA + (size_t)(m0 + (lane & 15)) * (size_t)K + ((lane >> 4) << 4);
  const bf16* rAb = Ab + zA + (size_t)(m0 + (lane & 15)) * (size_t)K + ((lane >> 4) << 4);
  const bf16* rBa = Ba + zB + (size_t)(n0 + (lane & 15)) * (size_t)K + ((lane >> 4) << 4);
  const bf16* rBb = Bb + zB + (size_t)(n0 + (lane & 15)) * (size_t)K + ((lane >> 4) << 4);

  v8f accAA = {}, accBB = {}, accAB = {}, accBA = {};
  for (int k0 = 0; k0 < K; k0 += 32) {
    if (k0 + 32 < K) {                               // global_prefetch_b8
      __builtin_prefetch(rBa + k0 + 32, 0, 1);
      __builtin_prefetch(rBb + k0 + 32, 0, 1);
    }
    v16bf fa = load_a_frag(rAa + k0);
    v16bf fb = load_a_frag(rAb + k0);
    v16bf ga = *(const v16bf*)(rBa + k0);
    v16bf gb = *(const v16bf*)(rBb + k0);
    accAA = wmma_bf16(fa, ga, accAA);
    accBB = wmma_bf16(fb, gb, accBB);
    accAB = wmma_bf16(fa, gb, accAB);
    accBA = wmma_bf16(fb, ga, accBA);
  }

  const int n = n0 + (lane & 15);
  const float bav = bias_a ? bias_a[n] : 0.0f;
  const float bbv = bias_b ? bias_b[n] : 0.0f;
#pragma unroll
  for (int r = 0; r < 8; ++r) {                      // C/D layout: m = r + 8*(lane/16)
    const int m = m0 + ((lane >> 4) << 3) + r;
    const size_t idx = zO + (size_t)m * (size_t)N + (size_t)n;
    float va = accAA[r] + coef * accBB[r] + bav;
    float vb = accAB[r] + accBA[r] + bbv;
    if (mode == 0) {
      ((float*)outA)[idx] = va;
      ((float*)outB)[idx] = vb;
    } else if (mode == 1) {
      ((unsigned short*)outA)[idx] = f32_bf16(va);
      ((unsigned short*)outB)[idx] = f32_bf16(vb);
    } else if (mode == 2) {                          // exact GELU, both components
      va = 0.5f * va * (1.0f + erff(va * 0.70710678f));
      vb = 0.5f * vb * (1.0f + erff(vb * 0.70710678f));
      ((unsigned short*)outA)[idx] = f32_bf16(va);
      ((unsigned short*)outB)[idx] = f32_bf16(vb);
    } else {                                          // magnitude * 1/sqrt(EMBED)
      ((float*)outA)[idx] = sqrtf(va * va + vb * vb) * oscale;
    }
  }
}

// Dual plain GEMM for attention*V: outA = W.Va, outB = W.Vb (V is KxN row-major)
__global__ __launch_bounds__(256) void attn_av_bf16(
    const bf16* __restrict__ W, const bf16* __restrict__ Va,
    const bf16* __restrict__ Vb,
    unsigned short* __restrict__ outA, unsigned short* __restrict__ outB,
    int M, int N, int K, long long sW, long long sV, long long sO) {
  const int lane = threadIdx.x & 31;
  const int wave = threadIdx.x >> 5;
  const int m0 = blockIdx.y * 64 + (wave & 3) * 16;
  const int n0 = blockIdx.x * 32 + (wave >> 2) * 16;
  const size_t zW = (size_t)blockIdx.z * (size_t)sW;
  const size_t zV = (size_t)blockIdx.z * (size_t)sV;
  const size_t zO = (size_t)blockIdx.z * (size_t)sO;

  const bf16* rW  = W  + zW + (size_t)(m0 + (lane & 15)) * (size_t)K + ((lane >> 4) << 4);
  const bf16* pVa = Va + zV + (size_t)(n0 + (lane & 15));
  const bf16* pVb = Vb + zV + (size_t)(n0 + (lane & 15));

  v8f aA = {}, aB = {};
  for (int k0 = 0; k0 < K; k0 += 32) {
    v16bf wf = load_a_frag(rW + k0);
    const int kb = k0 + ((lane >> 4) << 4);
    v16bf ba = load_b_strided(pVa + (size_t)kb * (size_t)N, N);
    v16bf bb = load_b_strided(pVb + (size_t)kb * (size_t)N, N);
    aA = wmma_bf16(wf, ba, aA);
    aB = wmma_bf16(wf, bb, aB);
  }
  const int n = n0 + (lane & 15);
#pragma unroll
  for (int r = 0; r < 8; ++r) {
    const int m = m0 + ((lane >> 4) << 3) + r;
    const size_t idx = zO + (size_t)m * (size_t)N + (size_t)n;
    outA[idx] = f32_bf16(aA[r]);
    outB[idx] = f32_bf16(aB[r]);
  }
}

__global__ void cvt_f32_to_bf16(const float* __restrict__ in,
                                unsigned short* __restrict__ out, size_t n) {
  size_t i = (size_t)blockIdx.x * blockDim.x + threadIdx.x;
  const size_t stride = (size_t)gridDim.x * blockDim.x;
  for (; i < n; i += stride) out[i] = f32_bf16(in[i]);
}

// Row softmax over S=2048 (one block/row), writes bf16 weights for next GEMM.
__global__ __launch_bounds__(256) void softmax_row_bf16(
    const float* __restrict__ s, unsigned short* __restrict__ w, int S) {
  __shared__ float sred[8];
  const float* p = s + (size_t)blockIdx.x * (size_t)S;
  unsigned short* q = w + (size_t)blockIdx.x * (size_t)S;
  float v[8];
  float mx = -3.402823466e38f;
#pragma unroll
  for (int i = 0; i < 8; ++i) {
    v[i] = p[threadIdx.x + i * 256];
    mx = fmaxf(mx, v[i]);
  }
  mx = bmax(mx, sred);
  float sum = 0.0f;
#pragma unroll
  for (int i = 0; i < 8; ++i) { v[i] = expf(v[i] - mx); sum += v[i]; }
  sum = bsum(sum, sred);
  const float inv = 1.0f / sum;
#pragma unroll
  for (int i = 0; i < 8; ++i) q[threadIdx.x + i * 256] = f32_bf16(v[i] * inv);
}

// out = LN(x + y) * g + b ; optionally also emit bf16 copy for next GEMM's A.
__global__ __launch_bounds__(256) void residual_ln(
    const float* __restrict__ x, const float* __restrict__ y,
    const float* __restrict__ g, const float* __restrict__ b,
    float* __restrict__ of, unsigned short* __restrict__ obf, int D) {
  __shared__ float sred[8];
  const size_t row = (size_t)blockIdx.x * (size_t)D;
  float v[4];
  float s = 0.0f;
#pragma unroll
  for (int i = 0; i < 4; ++i) {
    const int c = threadIdx.x + i * 256;
    v[i] = x[row + c] + y[row + c];
    s += v[i];
  }
  const float mean = bsum(s, sred) / (float)D;
  float qv = 0.0f;
#pragma unroll
  for (int i = 0; i < 4; ++i) { const float d = v[i] - mean; qv += d * d; }
  const float var = bsum(qv, sred) / (float)D;
  const float r = rsqrtf(var + 1e-6f);
#pragma unroll
  for (int i = 0; i < 4; ++i) {
    const int c = threadIdx.x + i * 256;
    const float o = (v[i] - mean) * r * g[c] + b[c];
    of[row + c] = o;
    if (obf) obf[row + c] = f32_bf16(o);
  }
}

// ---------------------------------------------------------------------------
extern "C" void kernel_launch(void* const* d_in, const int* in_sizes, int n_in,
                              void* d_out, int out_size, void* d_ws, size_t ws_size,
                              hipStream_t stream) {
  (void)in_sizes; (void)n_in; (void)out_size; (void)ws_size;
  const int B = 4, S = 2048, E = 1024, F = 4096;
  const size_t ACT  = (size_t)B * S * E;   // 8,388,608
  const size_t FACT = (size_t)B * S * F;   // 33,554,432
  const size_t SCR  = (size_t)B * S * S;   // 16,777,216

  // Flattened setup_inputs() dict order:
  // 0:xa 1:xb | q:2.wa 3.wb 4.ba 5.bb | k:6..9 | v:10..13 | o:14..17
  // ffn1:18.wa 19.wb 20.ba 21.bb | ffn2:22..25 | ln1:26.ga 27.bA 28.gb 29.bB
  // ln2:30..33
  auto in = [&](int i) { return (const float*)d_in[i]; };
  const float* xa = in(0);
  const float* xb = in(1);

  // ---- workspace arena (256B-aligned bump allocator, ~496MB) ----
  char* base = (char*)d_ws;
  size_t off = 0;
  auto alloc = [&](size_t bytes) -> void* {
    void* p = base + off;
    off = (off + bytes + 255) & ~(size_t)255;
    return p;
  };
  unsigned short* xa_bf = (unsigned short*)alloc(ACT * 2);
  unsigned short* xb_bf = (unsigned short*)alloc(ACT * 2);
  unsigned short* wq_a = (unsigned short*)alloc((size_t)E * E * 2);
  unsigned short* wq_b = (unsigned short*)alloc((size_t)E * E * 2);
  unsigned short* wk_a = (unsigned short*)alloc((size_t)E * E * 2);
  unsigned short* wk_b = (unsigned short*)alloc((size_t)E * E * 2);
  unsigned short* wv_a = (unsigned short*)alloc((size_t)E * E * 2);
  unsigned short* wv_b = (unsigned short*)alloc((size_t)E * E * 2);
  unsigned short* wo_a = (unsigned short*)alloc((size_t)E * E * 2);
  unsigned short* wo_b = (unsigned short*)alloc((size_t)E * E * 2);
  unsigned short* w1_a = (unsigned short*)alloc((size_t)F * E * 2);
  unsigned short* w1_b = (unsigned short*)alloc((size_t)F * E * 2);
  unsigned short* w2_a = (unsigned short*)alloc((size_t)E * F * 2);
  unsigned short* w2_b = (unsigned short*)alloc((size_t)E * F * 2);
  unsigned short* qa = (unsigned short*)alloc(ACT * 2);
  unsigned short* qb = (unsigned short*)alloc(ACT * 2);
  unsigned short* ka = (unsigned short*)alloc(ACT * 2);
  unsigned short* kb = (unsigned short*)alloc(ACT * 2);
  unsigned short* va = (unsigned short*)alloc(ACT * 2);
  unsigned short* vb = (unsigned short*)alloc(ACT * 2);
  float*          scores = (float*)alloc(SCR * 4);
  unsigned short* softw  = (unsigned short*)alloc(SCR * 2);
  float*          ha     = (float*)alloc(ACT * 4);
  float*          hb     = (float*)alloc(ACT * 4);
  unsigned short* ha_bf  = (unsigned short*)alloc(ACT * 2);
  unsigned short* hb_bf  = (unsigned short*)alloc(ACT * 2);
  unsigned short* g1a    = (unsigned short*)alloc(FACT * 2);
  unsigned short* g1b    = (unsigned short*)alloc(FACT * 2);
  // liveness-based aliases (dead buffers reused):
  unsigned short* attn_a = xa_bf;           // xa_bf dead after V GEMM
  unsigned short* attn_b = xb_bf;
  float* aa  = (float*)qa;                  // qa..qb dead after scores GEMM
  float* ab  = (float*)ka;                  // ka..kb dead after scores GEMM
  float* f2a = scores;                      // scores dead after softmax
  float* f2b = scores + ACT;

  const dim3 blk(256, 1, 1);
  auto cvt = [&](const float* src, unsigned short* dst, size_t n) {
    int grid = (int)((n + 2047) / 2048);
    if (grid > 8192) grid = 8192;
    cvt_f32_to_bf16<<<dim3(grid), blk, 0, stream>>>(src, dst, n);
  };
  cvt(xa, xa_bf, ACT);  cvt(xb, xb_bf, ACT);
  cvt(in(2),  wq_a, (size_t)E * E);  cvt(in(3),  wq_b, (size_t)E * E);
  cvt(in(6),  wk_a, (size_t)E * E);  cvt(in(7),  wk_b, (size_t)E * E);
  cvt(in(10), wv_a, (size_t)E * E);  cvt(in(11), wv_b, (size_t)E * E);
  cvt(in(14), wo_a, (size_t)E * E);  cvt(in(15), wo_b, (size_t)E * E);
  cvt(in(18), w1_a, (size_t)F * E);  cvt(in(19), w1_b, (size_t)F * E);
  cvt(in(22), w2_a, (size_t)E * F);  cvt(in(23), w2_b, (size_t)E * F);

  const float coef = -cosf(2.0f);  // j2 (theta=1) and the attention constant coincide
  auto qcg = [&](const void* Aa_, const void* Ab_, const void* Ba_, const void* Bb_,
                 const float* ba_, const float* bb_, void* oA, void* oB,
                 int M, int N, int K, long long sA, long long sB, long long sO,
                 int nz, float osc, int mode) {
    dim3 g((unsigned)(N / 32), (unsigned)(M / 64), (unsigned)nz);
    qc_gemm_bf16<<<g, blk, 0, stream>>>(
        (const bf16*)Aa_, (const bf16*)Ab_, (const bf16*)Ba_, (const bf16*)Bb_,
        ba_, bb_, oA, oB, M, N, K, sA, sB, sO, coef, osc, mode);
  };

  // Q/K/V projections (M = B*S rows, shared weights), bf16 outputs
  qcg(xa_bf, xb_bf, wq_a, wq_b, in(4),  in(5),  qa, qb, B * S, E, E, 0, 0, 0, 1, 1.0f, 1);
  qcg(xa_bf, xb_bf, wk_a, wk_b, in(8),  in(9),  ka, kb, B * S, E, E, 0, 0, 0, 1, 1.0f, 1);
  qcg(xa_bf, xb_bf, wv_a, wv_b, in(12), in(13), va, vb, B * S, E, E, 0, 0, 0, 1, 1.0f, 1);

  // Attention scores: magnitude(sa,sb) / sqrt(EMBED), batched over B
  qcg(qa, qb, ka, kb, nullptr, nullptr, scores, nullptr, S, S, E,
      (long long)S * E, (long long)S * E, (long long)S * S, B, 0.03125f, 3);

  softmax_row_bf16<<<dim3((unsigned)(B * S)), blk, 0, stream>>>(scores, softw, S);

  attn_av_bf16<<<dim3((unsigned)(E / 32), (unsigned)(S / 64), (unsigned)B), blk, 0, stream>>>(
      (const bf16*)softw, (const bf16*)va, (const bf16*)vb, attn_a, attn_b,
      S, E, S, (long long)S * S, (long long)S * E, (long long)S * E);

  // Output projection (f32 for residual)
  qcg(attn_a, attn_b, wo_a, wo_b, in(16), in(17), aa, ab, B * S, E, E, 0, 0, 0, 1, 1.0f, 0);

  // residual + LN1 (f32 ha/hb for residual-2, bf16 copies for FFN1 operands)
  residual_ln<<<dim3((unsigned)(B * S)), blk, 0, stream>>>(xa, aa, in(26), in(27), ha, ha_bf, E);
  residual_ln<<<dim3((unsigned)(B * S)), blk, 0, stream>>>(xb, ab, in(28), in(29), hb, hb_bf, E);

  // FFN1 with fused exact-GELU + bf16 epilogue
  qcg(ha_bf, hb_bf, w1_a, w1_b, in(20), in(21), g1a, g1b, B * S, F, E, 0, 0, 0, 1, 1.0f, 2);
  // FFN2 (f32 for residual)
  qcg(g1a, g1b, w2_a, w2_b, in(24), in(25), f2a, f2b, B * S, E, F, 0, 0, 0, 1, 1.0f, 0);

  // residual + LN2 straight into d_out = [2,B,S,D]
  float* out = (float*)d_out;
  residual_ln<<<dim3((unsigned)(B * S)), blk, 0, stream>>>(ha, f2a, in(30), in(31), out, nullptr, E);
  residual_ln<<<dim3((unsigned)(B * S)), blk, 0, stream>>>(hb, f2b, in(32), in(33), out + ACT, nullptr, E);
}
